// SFB_81088982549104
// MI455X (gfx1250) — matrix operand order repaired and checked by reading, not compiled, not run on hardware
//
#include <hip/hip_runtime.h>
#include <hip/hip_bf16.h>
#include <math.h>

typedef __bf16 bf16_t;
typedef __attribute__((ext_vector_type(16))) bf16_t bf16x16;
typedef __attribute__((ext_vector_type(8)))  float  f32x8;

constexpr int CH = 64;      // channels
constexpr int NP = 6400;    // 80*80 pixels
constexpr int NE = 1600;    // 40*40 event pixels
constexpr int BB = 2;       // batch
constexpr int NT = NP / 64; // number of 64-wide m-tiles

// ---------------- generic 1x1 conv + relu (fp32 -> fp32) ----------------
__global__ void k_conv_relu(const float* __restrict__ x, const float* __restrict__ w,
                            const float* __restrict__ bias, float* __restrict__ y, int npix) {
    int idx = blockIdx.x * blockDim.x + threadIdx.x;
    if (idx >= BB * npix) return;
    int b = idx / npix, n = idx - b * npix;
    size_t base = (size_t)b * CH * npix + n;
    float in[CH];
#pragma unroll
    for (int c = 0; c < CH; ++c) in[c] = x[base + (size_t)c * npix];
    for (int o = 0; o < CH; ++o) {
        float acc = bias[o];
#pragma unroll
        for (int c = 0; c < CH; ++c) acc += w[o * CH + c] * in[c];
        y[base + (size_t)o * npix] = fmaxf(acc, 0.f);
    }
}

// ---------------- x1 -> x2 -> {Q [N,C] bf16, V [C,N] bf16} ----------------
__global__ void k_x2_qv(const float* __restrict__ x1,
                        const float* __restrict__ wx1, const float* __restrict__ bx1,
                        const float* __restrict__ wq,  const float* __restrict__ bq,
                        const float* __restrict__ wv,  const float* __restrict__ bv,
                        bf16_t* __restrict__ Qbf, bf16_t* __restrict__ Vbf) {
    int idx = blockIdx.x * blockDim.x + threadIdx.x;
    if (idx >= BB * NP) return;
    int b = idx / NP, n = idx - b * NP;
    size_t base = (size_t)b * CH * NP + n;
    float in[CH], x2[CH];
#pragma unroll
    for (int c = 0; c < CH; ++c) in[c] = x1[base + (size_t)c * NP];
    for (int o = 0; o < CH; ++o) {
        float acc = bx1[o];
#pragma unroll
        for (int c = 0; c < CH; ++c) acc += wx1[o * CH + c] * in[c];
        x2[o] = fmaxf(acc, 0.f);
    }
    size_t qbase = ((size_t)b * NP + n) * CH;
    for (int o = 0; o < CH; ++o) {
        float acc = bq[o];
#pragma unroll
        for (int c = 0; c < CH; ++c) acc += wq[o * CH + c] * x2[c];
        Qbf[qbase + o] = (bf16_t)fmaxf(acc, 0.f);
    }
    for (int o = 0; o < CH; ++o) {
        float acc = bv[o];
#pragma unroll
        for (int c = 0; c < CH; ++c) acc += wv[o * CH + c] * x2[c];
        Vbf[base + (size_t)o * NP] = (bf16_t)fmaxf(acc, 0.f);
    }
}

// ------ ef0(40x40) --(nearest 2x fused)--> ef1 --> K [N,C] bf16 ------
__global__ void k_ef_k(const float* __restrict__ ef0,
                       const float* __restrict__ we1, const float* __restrict__ be1,
                       const float* __restrict__ wk,  const float* __restrict__ bk,
                       bf16_t* __restrict__ Kbf) {
    int idx = blockIdx.x * blockDim.x + threadIdx.x;
    if (idx >= BB * NP) return;
    int b = idx / NP, n = idx - b * NP;
    int h = n / 80, w = n - h * 80;
    int src = (h >> 1) * 40 + (w >> 1);
    size_t base = (size_t)b * CH * NE + src;
    float in[CH], e1[CH];
#pragma unroll
    for (int c = 0; c < CH; ++c) in[c] = ef0[base + (size_t)c * NE];
    for (int o = 0; o < CH; ++o) {
        float acc = be1[o];
#pragma unroll
        for (int c = 0; c < CH; ++c) acc += we1[o * CH + c] * in[c];
        e1[o] = fmaxf(acc, 0.f);
    }
    size_t kbase = ((size_t)b * NP + n) * CH;
    for (int o = 0; o < CH; ++o) {
        float acc = bk[o];
#pragma unroll
        for (int c = 0; c < CH; ++c) acc += wk[o * CH + c] * e1[c];
        Kbf[kbase + o] = (bf16_t)fmaxf(acc, 0.f);
    }
}

// ---------------- fused flash attention: softmax(Q^T K), out = V attn^T ----------------
// block = 128 threads (4 waves), 64 query rows per block; grid = (NP/64, B)
// software-pipelined: double-buffered K/V tiles in LDS, next-tile global loads
// issued at the top of the iteration, ds_store at the bottom (latency covered
// by the WMMA/softmax compute phase).
__launch_bounds__(128)
__global__ void k_attn(const bf16_t* __restrict__ Qbf, const bf16_t* __restrict__ Kbf,
                       const bf16_t* __restrict__ Vbf, const float* __restrict__ x1,
                       const float* __restrict__ gamma, float* __restrict__ out) {
    constexpr int LD = 72;  // padded row stride (bf16 elems): 144B, conflict-free
    __shared__ __align__(16) bf16_t sQ[64 * LD];
    __shared__ __align__(16) bf16_t sK[2][64 * LD];
    __shared__ __align__(16) bf16_t sV[2][64 * LD];
    __shared__ __align__(16) bf16_t sP[4 * 16 * LD];

    const int tid  = threadIdx.x;
    const int wave = tid >> 5, lane = tid & 31;
    const int half = lane >> 4, lj = lane & 15;
    const int b  = blockIdx.y;
    const int n0 = blockIdx.x * 64;

    // cooperative copy mapping: 64 rows x 8 chunks of 16B; part fixed per thread
    const int part = tid & 7;       // 16B chunk within a 128B row
    const int row0 = tid >> 3;      // rows row0, row0+16, row0+32, row0+48

    const bf16_t* Kbase = Kbf + (size_t)b * NP * CH;
    const bf16_t* Vbase = Vbf + (size_t)b * CH * NP;

    // stage Q tile [64 rows x 64 c] and first K/V tile into buffer 0
#pragma unroll
    for (int k = 0; k < 4; ++k) {
        int row = row0 + k * 16;
        *(uint4*)(sQ + row * LD + part * 8) =
            *(const uint4*)(Qbf + ((size_t)(b * NP + n0 + row)) * CH + part * 8);
        *(uint4*)(sK[0] + row * LD + part * 8) =
            *(const uint4*)(Kbase + (size_t)row * CH + part * 8);
        *(uint4*)(sV[0] + row * LD + part * 8) =
            *(const uint4*)(Vbase + (size_t)row * NP + part * 8);
    }
    __syncthreads();

    // A-fragments of this wave's 16 query rows (c 0..31 and c 32..63)
    bf16x16 qa0, qa1;
    {
        const bf16_t* qrow = sQ + (wave * 16 + lj) * LD;
        const int kb = half * 8;
        ((uint4*)&qa0)[0] = *(const uint4*)(qrow + kb);
        ((uint4*)&qa0)[1] = *(const uint4*)(qrow + kb + 16);
        ((uint4*)&qa1)[0] = *(const uint4*)(qrow + 32 + kb);
        ((uint4*)&qa1)[1] = *(const uint4*)(qrow + 32 + kb + 16);
    }

    f32x8 Oacc[4] = {};
    float mrow[8], lrow[8];
#pragma unroll
    for (int r = 0; r < 8; ++r) { mrow[r] = -INFINITY; lrow[r] = 0.f; }

    for (int mt = 0; mt < NT; ++mt) {
        const int cur = mt & 1, nxt = cur ^ 1;
        // prefetch next tile into registers (wraps to 0 on last iter: dead but branch-free)
        const int mnext = (mt + 1 < NT) ? (mt + 1) * 64 : 0;
        uint4 kr[4], vr[4];
#pragma unroll
        for (int k = 0; k < 4; ++k) {
            int row = row0 + k * 16;
            kr[k] = *(const uint4*)(Kbase + ((size_t)(mnext + row)) * CH + part * 8);
            vr[k] = *(const uint4*)(Vbase + (size_t)row * NP + mnext + part * 8);
        }

        // S = Q (16n x 64c) * K^T : four 16x16 m-subtiles, K-dim chained 2x32
        f32x8 S[4];
#pragma unroll
        for (int t = 0; t < 4; ++t) {
            const bf16_t* krow = sK[cur] + (t * 16 + lj) * LD;  // column m, contiguous c
            bf16x16 kb0, kb1;
            ((uint4*)&kb0)[0] = *(const uint4*)(krow + half * 8);
            ((uint4*)&kb0)[1] = *(const uint4*)(krow + half * 8 + 16);
            ((uint4*)&kb1)[0] = *(const uint4*)(krow + 32 + half * 8);
            ((uint4*)&kb1)[1] = *(const uint4*)(krow + 32 + half * 8 + 16);
            f32x8 acc = {};
            acc = __builtin_amdgcn_wmma_f32_16x16x32_bf16(false, qa0, false, kb0, (short)0, acc, false, false);
            acc = __builtin_amdgcn_wmma_f32_16x16x32_bf16(false, qa1, false, kb1, (short)0, acc, false, false);
            S[t] = acc;
        }

        // online softmax over this m-tile; stage P as bf16 A-layout source
#pragma unroll
        for (int r = 0; r < 8; ++r) {
            float mx = fmaxf(fmaxf(S[0][r], S[1][r]), fmaxf(S[2][r], S[3][r]));
            mx = fmaxf(mx, __shfl_xor(mx, 1, 16));
            mx = fmaxf(mx, __shfl_xor(mx, 2, 16));
            mx = fmaxf(mx, __shfl_xor(mx, 4, 16));
            mx = fmaxf(mx, __shfl_xor(mx, 8, 16));
            float mnew  = fmaxf(mrow[r], mx);
            float alpha = __expf(mrow[r] - mnew);
            mrow[r] = mnew;
            float rs = 0.f;
            bf16_t* prow = sP + (wave * 16 + r + half * 8) * LD;
#pragma unroll
            for (int t = 0; t < 4; ++t) {
                float pv = __expf(S[t][r] - mnew);
                rs += pv;
                prow[t * 16 + lj] = (bf16_t)pv;
            }
            rs += __shfl_xor(rs, 1, 16);
            rs += __shfl_xor(rs, 2, 16);
            rs += __shfl_xor(rs, 4, 16);
            rs += __shfl_xor(rs, 8, 16);
            lrow[r] = lrow[r] * alpha + rs;
#pragma unroll
            for (int ct = 0; ct < 4; ++ct) Oacc[ct][r] *= alpha;
        }
        __syncthreads();  // P visible (wave-crossing within its region is lane-crossing)

        // O += P (16n x 64m) * V^T (64m x 64ch)
        bf16x16 pa0, pa1;
        {
            const bf16_t* prow = sP + (wave * 16 + lj) * LD;
            const int kb = half * 8;
            ((uint4*)&pa0)[0] = *(const uint4*)(prow + kb);
            ((uint4*)&pa0)[1] = *(const uint4*)(prow + kb + 16);
            ((uint4*)&pa1)[0] = *(const uint4*)(prow + 32 + kb);
            ((uint4*)&pa1)[1] = *(const uint4*)(prow + 32 + kb + 16);
        }
#pragma unroll
        for (int ct = 0; ct < 4; ++ct) {
            const bf16_t* vrow = sV[cur] + (ct * 16 + lj) * LD;  // column ch, contiguous m
            bf16x16 vb0, vb1;
            ((uint4*)&vb0)[0] = *(const uint4*)(vrow + half * 8);
            ((uint4*)&vb0)[1] = *(const uint4*)(vrow + half * 8 + 16);
            ((uint4*)&vb1)[0] = *(const uint4*)(vrow + 32 + half * 8);
            ((uint4*)&vb1)[1] = *(const uint4*)(vrow + 32 + half * 8 + 16);
            Oacc[ct] = __builtin_amdgcn_wmma_f32_16x16x32_bf16(false, pa0, false, vb0, (short)0, Oacc[ct], false, false);
            Oacc[ct] = __builtin_amdgcn_wmma_f32_16x16x32_bf16(false, pa1, false, vb1, (short)0, Oacc[ct], false, false);
        }

        // commit prefetched tile into the other buffer; loadcnt wait lands here,
        // after the whole compute phase has covered the global latency
#pragma unroll
        for (int k = 0; k < 4; ++k) {
            int row = row0 + k * 16;
            *(uint4*)(sK[nxt] + row * LD + part * 8) = kr[k];
            *(uint4*)(sV[nxt] + row * LD + part * 8) = vr[k];
        }
        __syncthreads();  // next buffer visible; everyone done with sP and sK[cur]/sV[cur]
    }

    // epilogue: out = THITA * gamma * (O / l) + x1
    const float coef = 1e-4f * gamma[0];
    const int nb = n0 + wave * 16 + half * 8;  // 8 consecutive n per lane
#pragma unroll
    for (int ct = 0; ct < 4; ++ct) {
        int ch = ct * 16 + lj;
        size_t obase = ((size_t)b * CH + ch) * (size_t)NP + nb;
#pragma unroll
        for (int r = 0; r < 8; ++r)
            out[obase + r] = coef * (Oacc[ct][r] / lrow[r]) + x1[obase + r];
    }
}

extern "C" void kernel_launch(void* const* d_in, const int* in_sizes, int n_in,
                              void* d_out, int out_size, void* d_ws, size_t ws_size,
                              hipStream_t stream) {
    const float* x     = (const float*)d_in[0];
    const float* event = (const float*)d_in[1];
    const float* wx0 = (const float*)d_in[2];  const float* bx0 = (const float*)d_in[3];
    const float* wx1 = (const float*)d_in[4];  const float* bx1 = (const float*)d_in[5];
    const float* we0 = (const float*)d_in[6];  const float* be0 = (const float*)d_in[7];
    const float* we1 = (const float*)d_in[8];  const float* be1 = (const float*)d_in[9];
    const float* wq  = (const float*)d_in[10]; const float* bq  = (const float*)d_in[11];
    const float* wk  = (const float*)d_in[12]; const float* bk  = (const float*)d_in[13];
    const float* wv  = (const float*)d_in[14]; const float* bv  = (const float*)d_in[15];
    const float* gamma = (const float*)d_in[16];

    char* ws = (char*)d_ws;
    float*  x1  = (float*)(ws);                                  // 2*64*6400*4 = 3,276,800
    float*  ef0 = (float*)(ws + 3276800);                        // 2*64*1600*4 =   819,200
    bf16_t* Qbf = (bf16_t*)(ws + 3276800 + 819200);              // 1,638,400
    bf16_t* Kbf = (bf16_t*)(ws + 3276800 + 819200 + 1638400);    // 1,638,400
    bf16_t* Vbf = (bf16_t*)(ws + 3276800 + 819200 + 2 * 1638400);// 1,638,400 (total ~9.0 MB)

    k_conv_relu<<<dim3((BB * NP + 255) / 256), 256, 0, stream>>>(x, wx0, bx0, x1, NP);
    k_conv_relu<<<dim3((BB * NE + 255) / 256), 256, 0, stream>>>(event, we0, be0, ef0, NE);
    k_x2_qv<<<dim3((BB * NP + 255) / 256), 256, 0, stream>>>(x1, wx1, bx1, wq, bq, wv, bv, Qbf, Vbf);
    k_ef_k<<<dim3((BB * NP + 255) / 256), 256, 0, stream>>>(ef0, we1, be1, wk, bk, Kbf);
    k_attn<<<dim3(NP / 64, BB), 128, 0, stream>>>(Qbf, Kbf, Vbf, x1, gamma, (float*)d_out);
}
